// CSPLayer_49641232007336
// MI455X (gfx1250) — matrix operand order repaired and checked by reading, not compiled, not run on hardware
//
#include <hip/hip_runtime.h>

typedef __attribute__((ext_vector_type(16))) __bf16 bf16x16;
typedef __attribute__((ext_vector_type(8)))  float  v8f;
typedef __attribute__((ext_vector_type(8)))  unsigned short ushort8;
typedef __attribute__((ext_vector_type(16))) unsigned short ushort16;
typedef __attribute__((ext_vector_type(4)))  int    i32x4;

#define DEV_INLINE __device__ __forceinline__

constexpr int NN = 50000;   // nodes
constexpr int EE = 200000;  // edges
constexpr int GG = 2000;    // graphs

static_assert(sizeof(bf16x16) == 32, "bf16x16 must be 32B");

// ---- CDNA5 async global->LDS copy support (guarded; falls back to reg copy) ----
#if defined(__has_builtin)
#if __has_builtin(__builtin_amdgcn_global_load_async_to_lds_b128)
#define HAVE_ASYNC_COPY 1
#endif
#endif
#ifndef HAVE_ASYNC_COPY
#define HAVE_ASYNC_COPY 0
#endif

#define AS_GLOBAL __attribute__((address_space(1)))
#define AS_LDS    __attribute__((address_space(3)))

DEV_INLINE void async_wait0() {
#if HAVE_ASYNC_COPY
#if __has_builtin(__builtin_amdgcn_s_wait_asynccnt)
  __builtin_amdgcn_s_wait_asynccnt(0);
#else
  asm volatile("s_wait_asynccnt 0" ::: "memory");
#endif
#endif
}

// Copy one 16KB B chunk (64B per thread, 256 threads) global -> LDS.
DEV_INLINE void stage_chunk(unsigned short* dstLds, const unsigned short* srcGlob, int tid) {
#if HAVE_ASYNC_COPY
  AS_GLOBAL i32x4* g =
      (AS_GLOBAL i32x4*)(unsigned long long)(const void*)(srcGlob + tid * 32);
  AS_LDS i32x4* l =
      (AS_LDS i32x4*)(unsigned)(unsigned long long)(void*)(dstLds + tid * 32);
  // instruction offset applies to both LDS and global addresses (ISA 08 / 4.4)
  __builtin_amdgcn_global_load_async_to_lds_b128(g, l, 0, 0);
  __builtin_amdgcn_global_load_async_to_lds_b128(g, l, 16, 0);
  __builtin_amdgcn_global_load_async_to_lds_b128(g, l, 32, 0);
  __builtin_amdgcn_global_load_async_to_lds_b128(g, l, 48, 0);
#else
  const ushort8* gp = (const ushort8*)srcGlob + tid * 4;
  ushort8* sp = (ushort8*)(dstLds + tid * 32);
  ushort8 a0 = gp[0], a1 = gp[1], a2 = gp[2], a3 = gp[3];
  sp[0] = a0; sp[1] = a1; sp[2] = a2; sp[3] = a3;
#endif
}

DEV_INLINE unsigned short bf16bits(float f) {
  unsigned u = __builtin_bit_cast(unsigned, f);
  u += 0x7FFFu + ((u >> 16) & 1u);   // round-to-nearest-even
  return (unsigned short)(u >> 16);
}

DEV_INLINE float silu_f(float x) { return x / (1.f + __expf(-x)); }

// Load a 16-bf16 A fragment: two contiguous 16B runs (K runs [8h,8h+8) and [16+8h,16+8h+8))
DEV_INLINE ushort16 ld_frag(const unsigned short* p) {
  ushort8 lo = *(const ushort8*)p;
  ushort8 hi = *(const ushort8*)(p + 16);
  return __builtin_shufflevector(lo, hi, 0,1,2,3,4,5,6,7,8,9,10,11,12,13,14,15);
}

// ---------------------------------------------------------------- layernorm -> bf16
__global__ void k_layernorm(const float* __restrict__ h,
                            const float* __restrict__ lnw,
                            const float* __restrict__ lnb,
                            unsigned short* __restrict__ hnorm) {
  const int lane = threadIdx.x & 31;
  const int row  = blockIdx.x * 8 + (threadIdx.x >> 5);
  const float* hp = h + (size_t)row * 256;
  float v[8], s = 0.f, s2 = 0.f;
#pragma unroll
  for (int i = 0; i < 8; ++i) { v[i] = hp[lane + 32 * i]; s += v[i]; s2 += v[i] * v[i]; }
#pragma unroll
  for (int m = 16; m >= 1; m >>= 1) { s += __shfl_xor(s, m, 32); s2 += __shfl_xor(s2, m, 32); }
  const float mu   = s * (1.f / 256.f);
  const float var  = s2 * (1.f / 256.f) - mu * mu;
  const float rstd = rsqrtf(var + 1e-5f);
#pragma unroll
  for (int i = 0; i < 8; ++i) {
    const int col = lane + 32 * i;
    hnorm[(size_t)row * 256 + col] = bf16bits((v[i] - mu) * rstd * lnw[col] + lnb[col]);
  }
}

// ---------------------------------------------------------------- per-graph prep
// ltl[g] = L L^T ; gbias[g] = b_e1 + lf@W_e1[515:521] + (na@W_na)@W_e1[1289:1545]
__global__ void k_graphprep(const float* __restrict__ lf,
                            const float* __restrict__ L,
                            const float* __restrict__ na,
                            const float* __restrict__ Wna,
                            const float* __restrict__ We1,
                            const float* __restrict__ be1,
                            float* __restrict__ ltl,
                            float* __restrict__ gbias) {
  __shared__ float naS[100];
  __shared__ float ne[256];
  const int g = blockIdx.x;
  const int t = threadIdx.x;
  if (t < 100) naS[t] = na[g * 100 + t];
  if (t < 9) {
    const int i = t / 3, j = t % 3;
    const float* Lg = L + g * 9;
    ltl[g * 9 + t] = Lg[i*3+0]*Lg[j*3+0] + Lg[i*3+1]*Lg[j*3+1] + Lg[i*3+2]*Lg[j*3+2];
  }
  __syncthreads();
  float s = 0.f;
  for (int a = 0; a < 100; ++a) s += naS[a] * Wna[a * 256 + t];
  ne[t] = s;
  __syncthreads();
  float gv = be1[t];
  const float* lfg = lf + g * 6;
#pragma unroll
  for (int d = 0; d < 6; ++d) gv += lfg[d] * We1[(size_t)(515 + d) * 256 + t];
  for (int k = 0; k < 256; ++k) gv += ne[k] * We1[(size_t)(1289 + k) * 256 + t];
  gbias[(size_t)g * 256 + t] = gv;
}

// ---------------------------------------------------------------- weight packing
// dst layout [chunk][ntile(16)][lane(32)][j(16)] bf16; B frag: col=lane&15, h=lane>>4, K=8h+j (j<8), K=16+8h+(j-8) (j>=8)
__global__ void k_pack(const float* __restrict__ W, unsigned short* __restrict__ dst,
                       int rowBase, int validRows, int total) {
  const int idx = blockIdx.x * 256 + threadIdx.x;
  if (idx >= total) return;
  const int j     = idx & 15;
  const int lane  = (idx >> 4) & 31;
  const int ntile = (idx >> 9) & 15;
  const int c     = idx >> 13;
  const int k     = 32 * c + ((lane & 16) ? 8 : 0) + (j < 8 ? j : j + 8);
  const int col   = ntile * 16 + (lane & 15);
  const float v   = (k < validRows) ? W[(size_t)(rowBase + k) * 256 + col] : 0.f;
  dst[idx] = bf16bits(v);
}

// ---------------------------------------------------------------- zero scratch
__global__ void k_zero(float* __restrict__ esum, float* __restrict__ cnt) {
  const size_t i = (size_t)blockIdx.x * 256 + threadIdx.x;
  if (i < (size_t)NN * 256) esum[i] = 0.f;
  if (i < NN) cnt[i] = 0.f;
}

// ---------------------------------------------------------------- edge MLP (the big one)
// block = 256 thr = 8 waves; 64 edges/block; wave (mtile=w>>1) x (nhalf=w&1) -> 16x128 slab
__global__ __launch_bounds__(256, 2) void k_edge(
    const unsigned short* __restrict__ hnorm,
    const int*   __restrict__ eidx,
    const int*   __restrict__ e2g,
    const float* __restrict__ fdiff,
    const float* __restrict__ ltl,
    const float* __restrict__ gbias,
    const unsigned short* __restrict__ Bp1,
    const unsigned short* __restrict__ Bp2,
    const float* __restrict__ be2,
    float* __restrict__ esum,
    float* __restrict__ cnt) {
  __shared__ __attribute__((aligned(32))) unsigned short Bst[2][8192];  // B chunk double buffer
  __shared__ __attribute__((aligned(16))) unsigned short Z1[64][264];   // silu(z1) transpose tile
  __shared__ int Seg[64];
  __shared__ int Gid[64];

  const int tid   = threadIdx.x;
  const int lane  = tid & 31;
  const int wave  = tid >> 5;
  const int mtile = wave >> 1;
  const int nhalf = wave & 1;
  const int hh    = lane >> 4;     // K-half of this lane
  const int mrow  = lane & 15;     // row within M-tile / col within N-tile
  const int lrow  = mtile * 16 + mrow;
  const int edge  = blockIdx.x * 64 + lrow;  // E % 64 == 0

  const int src = eidx[edge];       // edge_index[0]: hi gather + scatter segment
  const int dst = eidx[EE + edge];  // edge_index[1]: hj gather
  const int g   = e2g[edge];
  const float fd0 = fdiff[edge * 3 + 0];
  const float fd1 = fdiff[edge * 3 + 1];
  const float fd2 = fdiff[edge * 3 + 2];

  const float* M = ltl + g * 9;
  const float d0 = M[0]*fd0 + M[1]*fd1 + M[2]*fd2;
  const float d1 = M[3]*fd0 + M[4]*fd1 + M[5]*fd2;
  const float d2 = M[6]*fd0 + M[7]*fd1 + M[8]*fd2;
  const float inv = 1.f / (sqrtf(d0*d0 + d1*d1 + d2*d2) + 1e-12f);
  const float ud0 = d0 * inv, ud1 = d1 * inv, ud2 = d2 * inv;

  if (nhalf == 0 && hh == 0) { Seg[lrow] = src; Gid[lrow] = g; }

  const unsigned short* hip8 = hnorm + (size_t)src * 256;
  const unsigned short* hjp8 = hnorm + (size_t)dst * 256;

  v8f acc[8];
#pragma unroll
  for (int nt = 0; nt < 8; ++nt) acc[nt] = (v8f)0.f;

  stage_chunk(&Bst[0][0], Bp1, tid);
  async_wait0();
  __syncthreads();

  // ---- GEMM1: chunks 0-7 = hi, 8-15 = hj, 16-39 = sinusoidal emb, 40 = unit_dots (padded)
  for (int c = 0; c < 41; ++c) {
    if (c + 1 < 41) stage_chunk(&Bst[(c + 1) & 1][0], Bp1 + (size_t)(c + 1) * 8192, tid);
    ushort16 au;
    if (c < 8) {
      au = ld_frag(hip8 + 32 * c + 8 * hh);
    } else if (c < 16) {
      au = ld_frag(hjp8 + 32 * (c - 8) + 8 * hh);
    } else if (c < 40) {
      const int kbase = 32 * (c - 16);
#pragma unroll
      for (int j = 0; j < 16; ++j) {
        const int k    = 8 * hh + (j < 8 ? j : j + 8);
        const int colv = kbase + k;                 // 0..767 within dis_emb
        const bool isc = colv >= 384;
        const int cc   = isc ? colv - 384 : colv;
        const int dsel = cc >> 7;
        const int f    = cc & 127;
        const float fdd = (dsel == 0) ? fd0 : ((dsel == 1) ? fd1 : fd2);
        const float x   = fdd * (6.28318530717958647f * (float)f);
        au[j] = bf16bits(isc ? __cosf(x) : __sinf(x));
      }
    } else {  // unit_dots chunk: K=0..2 live, rest zero
#pragma unroll
      for (int j = 0; j < 16; ++j) {
        const int k = 8 * hh + (j < 8 ? j : j + 8);
        au[j] = bf16bits((k == 0) ? ud0 : (k == 1) ? ud1 : (k == 2) ? ud2 : 0.f);
      }
    }
    const bf16x16 a = __builtin_bit_cast(bf16x16, au);
    const unsigned short* bbase = &Bst[c & 1][0];
#pragma unroll
    for (int nt = 0; nt < 8; ++nt) {
      const int ntile = nhalf * 8 + nt;
      const ushort16 bu = *(const ushort16*)(bbase + (ntile * 32 + lane) * 16);
      const bf16x16 b = __builtin_bit_cast(bf16x16, bu);
      acc[nt] = __builtin_amdgcn_wmma_f32_16x16x32_bf16(false, a, false, b, (short)0,
                                                        acc[nt], false, false);
    }
    async_wait0();
    __syncthreads();
  }

  // epilogue 1: add per-graph bias, silu, transpose into LDS as bf16
#pragma unroll
  for (int nt = 0; nt < 8; ++nt) {
    const int col = (nhalf * 8 + nt) * 16 + mrow;
#pragma unroll
    for (int r = 0; r < 8; ++r) {
      const int lr = mtile * 16 + r + 8 * hh;
      const float z = acc[nt][r] + gbias[(size_t)Gid[lr] * 256 + col];
      Z1[lr][col] = bf16bits(silu_f(z));
    }
  }
  __syncthreads();

  // ---- GEMM2: z1 @ W_e2 (K=256, 8 chunks)
#pragma unroll
  for (int nt = 0; nt < 8; ++nt) acc[nt] = (v8f)0.f;
  stage_chunk(&Bst[0][0], Bp2, tid);
  async_wait0();
  __syncthreads();
  const unsigned short* zrow = &Z1[mtile * 16 + mrow][0];
  for (int c = 0; c < 8; ++c) {
    if (c + 1 < 8) stage_chunk(&Bst[(c + 1) & 1][0], Bp2 + (size_t)(c + 1) * 8192, tid);
    const ushort16 au = ld_frag(zrow + 32 * c + 8 * hh);
    const bf16x16 a = __builtin_bit_cast(bf16x16, au);
    const unsigned short* bbase = &Bst[c & 1][0];
#pragma unroll
    for (int nt = 0; nt < 8; ++nt) {
      const int ntile = nhalf * 8 + nt;
      const ushort16 bu = *(const ushort16*)(bbase + (ntile * 32 + lane) * 16);
      const bf16x16 b = __builtin_bit_cast(bf16x16, bu);
      acc[nt] = __builtin_amdgcn_wmma_f32_16x16x32_bf16(false, a, false, b, (short)0,
                                                        acc[nt], false, false);
    }
    async_wait0();
    __syncthreads();
  }

  // epilogue 2: silu and scatter-add into segment sums (L2 atomics)
#pragma unroll
  for (int nt = 0; nt < 8; ++nt) {
    const int col = (nhalf * 8 + nt) * 16 + mrow;
    const float bb = be2[col];
#pragma unroll
    for (int r = 0; r < 8; ++r) {
      const int lr = mtile * 16 + r + 8 * hh;
      const float ev = silu_f(acc[nt][r] + bb);
      atomicAdd(&esum[(size_t)Seg[lr] * 256 + col], ev);
    }
  }
  if (nhalf == 0 && mrow == 0) {
#pragma unroll
    for (int r = 0; r < 8; ++r) atomicAdd(&cnt[Seg[mtile * 16 + r + 8 * hh]], 1.0f);
  }
}

// ---------------------------------------------------------------- agg = esum / max(cnt,1) -> bf16
__global__ void k_agg(const float* __restrict__ esum, const float* __restrict__ cnt,
                      unsigned short* __restrict__ aggb) {
  const size_t i = (size_t)blockIdx.x * 256 + threadIdx.x;
  if (i >= (size_t)NN * 256) return;
  aggb[i] = bf16bits(esum[i] / fmaxf(cnt[i >> 8], 1.f));
}

// ---------------------------------------------------------------- node MLP + residual
__global__ __launch_bounds__(256, 2) void k_node(
    const unsigned short* __restrict__ hnorm,
    const unsigned short* __restrict__ aggb,
    const float* __restrict__ hin,
    const unsigned short* __restrict__ BpN1,
    const unsigned short* __restrict__ BpN2,
    const float* __restrict__ bn1,
    const float* __restrict__ bn2,
    float* __restrict__ out) {
  __shared__ __attribute__((aligned(32))) unsigned short Bst[2][8192];
  __shared__ __attribute__((aligned(16))) unsigned short Z1[64][264];

  const int tid   = threadIdx.x;
  const int lane  = tid & 31;
  const int wave  = tid >> 5;
  const int mtile = wave >> 1;
  const int nhalf = wave & 1;
  const int hh    = lane >> 4;
  const int mrow  = lane & 15;
  const int lrow  = mtile * 16 + mrow;
  const int node  = blockIdx.x * 64 + lrow;
  const int cn    = node < NN ? node : NN - 1;

  const unsigned short* hp = hnorm + (size_t)cn * 256;
  const unsigned short* ap = aggb + (size_t)cn * 256;

  v8f acc[8];
#pragma unroll
  for (int nt = 0; nt < 8; ++nt) acc[nt] = (v8f)0.f;
  stage_chunk(&Bst[0][0], BpN1, tid);
  async_wait0();
  __syncthreads();

  // GEMM1: K=512 -> chunks 0-7 from h_norm, 8-15 from agg
  for (int c = 0; c < 16; ++c) {
    if (c + 1 < 16) stage_chunk(&Bst[(c + 1) & 1][0], BpN1 + (size_t)(c + 1) * 8192, tid);
    const ushort16 au = (c < 8) ? ld_frag(hp + 32 * c + 8 * hh)
                                : ld_frag(ap + 32 * (c - 8) + 8 * hh);
    const bf16x16 a = __builtin_bit_cast(bf16x16, au);
    const unsigned short* bbase = &Bst[c & 1][0];
#pragma unroll
    for (int nt = 0; nt < 8; ++nt) {
      const int ntile = nhalf * 8 + nt;
      const ushort16 bu = *(const ushort16*)(bbase + (ntile * 32 + lane) * 16);
      const bf16x16 b = __builtin_bit_cast(bf16x16, bu);
      acc[nt] = __builtin_amdgcn_wmma_f32_16x16x32_bf16(false, a, false, b, (short)0,
                                                        acc[nt], false, false);
    }
    async_wait0();
    __syncthreads();
  }

#pragma unroll
  for (int nt = 0; nt < 8; ++nt) {
    const int col = (nhalf * 8 + nt) * 16 + mrow;
#pragma unroll
    for (int r = 0; r < 8; ++r) {
      const int lr = mtile * 16 + r + 8 * hh;
      Z1[lr][col] = bf16bits(silu_f(acc[nt][r] + bn1[col]));
    }
  }
  __syncthreads();

  // GEMM2: K=256
#pragma unroll
  for (int nt = 0; nt < 8; ++nt) acc[nt] = (v8f)0.f;
  stage_chunk(&Bst[0][0], BpN2, tid);
  async_wait0();
  __syncthreads();
  const unsigned short* zrow = &Z1[mtile * 16 + mrow][0];
  for (int c = 0; c < 8; ++c) {
    if (c + 1 < 8) stage_chunk(&Bst[(c + 1) & 1][0], BpN2 + (size_t)(c + 1) * 8192, tid);
    const ushort16 au = ld_frag(zrow + 32 * c + 8 * hh);
    const bf16x16 a = __builtin_bit_cast(bf16x16, au);
    const unsigned short* bbase = &Bst[c & 1][0];
#pragma unroll
    for (int nt = 0; nt < 8; ++nt) {
      const int ntile = nhalf * 8 + nt;
      const ushort16 bu = *(const ushort16*)(bbase + (ntile * 32 + lane) * 16);
      const bf16x16 b = __builtin_bit_cast(bf16x16, bu);
      acc[nt] = __builtin_amdgcn_wmma_f32_16x16x32_bf16(false, a, false, b, (short)0,
                                                        acc[nt], false, false);
    }
    async_wait0();
    __syncthreads();
  }

#pragma unroll
  for (int nt = 0; nt < 8; ++nt) {
    const int col = (nhalf * 8 + nt) * 16 + mrow;
    const float bb = bn2[col];
#pragma unroll
    for (int r = 0; r < 8; ++r) {
      const int lr = mtile * 16 + r + 8 * hh;
      const int nrow = blockIdx.x * 64 + lr;
      if (nrow < NN) {
        const float o = silu_f(acc[nt][r] + bb);
        out[(size_t)nrow * 256 + col] = hin[(size_t)nrow * 256 + col] + o;
      }
    }
  }
}

// ---------------------------------------------------------------- launcher
extern "C" void kernel_launch(void* const* d_in, const int* in_sizes, int n_in,
                              void* d_out, int out_size, void* d_ws, size_t ws_size,
                              hipStream_t stream) {
  (void)in_sizes; (void)n_in; (void)out_size; (void)ws_size;
  const float* h     = (const float*)d_in[0];
  const float* lf    = (const float*)d_in[1];
  const float* L     = (const float*)d_in[2];
  const int*   eidx  = (const int*)d_in[3];
  const int*   e2g   = (const int*)d_in[4];
  const float* fdiff = (const float*)d_in[5];
  const float* na    = (const float*)d_in[6];
  const float* Wna   = (const float*)d_in[7];
  const float* We1   = (const float*)d_in[8];
  const float* be1   = (const float*)d_in[9];
  const float* We2   = (const float*)d_in[10];
  const float* be2   = (const float*)d_in[11];
  const float* Wn1   = (const float*)d_in[12];
  const float* bn1   = (const float*)d_in[13];
  const float* Wn2   = (const float*)d_in[14];
  const float* bn2   = (const float*)d_in[15];
  const float* lnw   = (const float*)d_in[16];
  const float* lnb   = (const float*)d_in[17];
  float* out = (float*)d_out;

  char* ws = (char*)d_ws;
  size_t off = 0;
  auto take = [&](size_t bytes) -> char* {
    char* p = ws + off;
    off = (off + bytes + 255) & ~(size_t)255;
    return p;
  };
  unsigned short* hnorm = (unsigned short*)take((size_t)NN * 256 * 2);
  unsigned short* aggb  = (unsigned short*)take((size_t)NN * 256 * 2);
  float* esum  = (float*)take((size_t)NN * 256 * 4);
  float* cnt   = (float*)take((size_t)NN * 4);
  float* gbias = (float*)take((size_t)GG * 256 * 4);
  float* ltl   = (float*)take((size_t)GG * 9 * 4);
  unsigned short* Bp1  = (unsigned short*)take((size_t)41 * 8192 * 2);
  unsigned short* Bp2  = (unsigned short*)take((size_t)8 * 8192 * 2);
  unsigned short* BpN1 = (unsigned short*)take((size_t)16 * 8192 * 2);
  unsigned short* BpN2 = (unsigned short*)take((size_t)8 * 8192 * 2);

  k_zero<<<dim3((NN * 256 + 255) / 256), dim3(256), 0, stream>>>(esum, cnt);
  k_layernorm<<<dim3(NN / 8), dim3(256), 0, stream>>>(h, lnw, lnb, hnorm);
  k_graphprep<<<dim3(GG), dim3(256), 0, stream>>>(lf, L, na, Wna, We1, be1, ltl, gbias);

  auto pack = [&](const float* W, unsigned short* dst, int rowBase, int valid, int nChunks) {
    const int total = nChunks * 8192;
    k_pack<<<dim3((total + 255) / 256), dim3(256), 0, stream>>>(W, dst, rowBase, valid, total);
  };
  pack(We1, Bp1 + (size_t)0 * 8192, 3, 256, 8);     // hi rows
  pack(We1, Bp1 + (size_t)8 * 8192, 259, 256, 8);   // hj rows
  pack(We1, Bp1 + (size_t)16 * 8192, 521, 768, 24); // dis_emb rows
  pack(We1, Bp1 + (size_t)40 * 8192, 0, 3, 1);      // unit_dots rows (zero padded)
  pack(We2, Bp2, 0, 256, 8);
  pack(Wn1, BpN1, 0, 512, 16);
  pack(Wn2, BpN2, 0, 256, 8);

  k_edge<<<dim3(EE / 64), dim3(256), 0, stream>>>(hnorm, eidx, e2g, fdiff, ltl, gbias,
                                                  Bp1, Bp2, be2, esum, cnt);
  k_agg<<<dim3((NN * 256 + 255) / 256), dim3(256), 0, stream>>>(esum, cnt, aggb);
  k_node<<<dim3((NN + 63) / 64), dim3(256), 0, stream>>>(hnorm, aggb, h, BpN1, BpN2,
                                                         bn1, bn2, out);
}
